// PyramidViG_82420422410487
// MI455X (gfx1250) — matrix operand order repaired and checked by reading, not compiled, not run on hardware
//
#include <hip/hip_runtime.h>
#include <math.h>

// ---------------------------------------------------------------------------
// Pyramid ViG forward on MI455X (gfx1250, wave32).
// All dense 1x1-conv GEMMs and the KNN similarity Gram matrix run on
// v_wmma_f32_16x16x32_bf16 (bf16 in, fp32 accumulate). Activations are fp32
// NHWC in workspace; fp32->bf16 conversion happens while staging tiles to LDS
// with vectorized b128 traffic. B operand is always (N,K) row-major (torch
// conv weight layout / nf rows), staged transposed into LDS so both global
// reads and fragment reads are contiguous. The K-loop is register-pipelined:
// next tile's global_load_b128s are issued before the wmma section so they
// overlap matrix compute; bounds are handled by address clamp + cndmask zero
// (no exec-mask branches in the hot loop).
// ---------------------------------------------------------------------------

typedef __attribute__((ext_vector_type(16))) __bf16 v16bf;
typedef __attribute__((ext_vector_type(8)))  __bf16 v8bf;
typedef __attribute__((ext_vector_type(4)))  __bf16 v4bf;
typedef __attribute__((ext_vector_type(8)))  float  v8f;

#define TBM 64
#define TBN 64
#define TBK 32
#define LDT 40   // padded LDS row stride (elements); 80B = multiple of 16B

static __device__ __forceinline__ float gelu_erf(float x) {
  return 0.5f * x * (1.0f + erff(x * 0.70710678118f));
}

// C[M,N] = act( BN( A[M,K] * B^T + bias ) ) (+ residual)
// A: (M,K) fp32 row-major.  Bm: (N,K) fp32 row-major (read as B^T).
__global__ __launch_bounds__(256) void gemm_wmma_bn_kernel(
    const float* __restrict__ A, const float* __restrict__ Bm, float* __restrict__ C,
    int M, int N, int K,
    const float* __restrict__ bias, const float* __restrict__ gamma,
    const float* __restrict__ beta, const float* __restrict__ residual, int do_gelu)
{
  __shared__ alignas(16) __bf16 As[TBM][LDT];   // [m][k]
  __shared__ alignas(16) __bf16 Bs[TBN][LDT];   // [n][k]  (B transposed)

  const int tid  = threadIdx.x;
  const int wid  = tid >> 5;
  const int lane = tid & 31;
  const int wr   = wid & 3;        // 4 wave rows  x16 = 64 M
  const int wc   = wid >> 2;       // 2 wave cols  x32 = 64 N (2 subtiles each)
  const int m0   = blockIdx.x * TBM;
  const int n0   = blockIdx.y * TBN;
  const int half = lane >> 4;
  const int l16  = lane & 15;

  v8f acc0 = {}, acc1 = {};

  // staging coordinates: 8 threads per row, float4 each -> 32 K per row
  const int srow = tid >> 3;          // 0..31
  const int skk  = (tid & 7) * 4;     // 0,4,...,28

  float4 ra[2], rb[2];

  // unconditional clamped loads; out-of-range lanes zero-selected afterwards.
  auto prefetch = [&](int k0) {
    const int gk   = k0 + skk;
    const bool kin = (gk < K);        // K multiple of 16 -> float4 all-in or all-out
    const int gkc  = kin ? gk : 0;
#pragma unroll
    for (int p = 0; p < 2; p++) {
      const int r  = srow + p * 32;
      const int gm = m0 + r;
      const int gn = n0 + r;
      const int gmc = (gm < M) ? gm : (M - 1);
      const int gnc = (gn < N) ? gn : (N - 1);
      float4 fa = *(const float4*)(A  + (size_t)gmc * K + gkc);
      float4 fb = *(const float4*)(Bm + (size_t)gnc * K + gkc);
      const bool oa = kin && (gm < M);
      const bool ob = kin && (gn < N);
      if (!oa) { fa.x = 0.f; fa.y = 0.f; fa.z = 0.f; fa.w = 0.f; }
      if (!ob) { fb.x = 0.f; fb.y = 0.f; fb.z = 0.f; fb.w = 0.f; }
      ra[p] = fa;
      rb[p] = fb;
    }
  };

  prefetch(0);

  for (int k0 = 0; k0 < K; k0 += TBK) {
    // convert staged registers to bf16 and store to LDS
#pragma unroll
    for (int p = 0; p < 2; p++) {
      const int r = srow + p * 32;
      v4bf oa, ob;
      oa[0] = (__bf16)ra[p].x; oa[1] = (__bf16)ra[p].y;
      oa[2] = (__bf16)ra[p].z; oa[3] = (__bf16)ra[p].w;
      ob[0] = (__bf16)rb[p].x; ob[1] = (__bf16)rb[p].y;
      ob[2] = (__bf16)rb[p].z; ob[3] = (__bf16)rb[p].w;
      *(v4bf*)&As[r][skk] = oa;
      *(v4bf*)&Bs[r][skk] = ob;
    }
    __syncthreads();

    // issue next tile's global loads; they overlap the wmma section below
    if (k0 + TBK < K) prefetch(k0 + TBK);

    // A fragment: lane row = wr*16+l16; K = 16*half + 0..15 (contiguous)
    const int arow = wr * 16 + l16;
    v8bf alo = *(const v8bf*)&As[arow][16 * half];
    v8bf ahi = *(const v8bf*)&As[arow][16 * half + 8];
    v16bf a = __builtin_shufflevector(alo, ahi, 0, 1, 2, 3, 4, 5, 6, 7,
                                      8, 9, 10, 11, 12, 13, 14, 15);
    // B fragments: lane col = base+l16; K = 16*half + 0..15 (contiguous)
    {
      const int brow = wc * 32 + l16;
      v8bf blo = *(const v8bf*)&Bs[brow][16 * half];
      v8bf bhi = *(const v8bf*)&Bs[brow][16 * half + 8];
      v16bf b = __builtin_shufflevector(blo, bhi, 0, 1, 2, 3, 4, 5, 6, 7,
                                        8, 9, 10, 11, 12, 13, 14, 15);
      acc0 = __builtin_amdgcn_wmma_f32_16x16x32_bf16(
          false, a, false, b, (short)0, acc0, false, false);
    }
    {
      const int brow = wc * 32 + 16 + l16;
      v8bf blo = *(const v8bf*)&Bs[brow][16 * half];
      v8bf bhi = *(const v8bf*)&Bs[brow][16 * half + 8];
      v16bf b = __builtin_shufflevector(blo, bhi, 0, 1, 2, 3, 4, 5, 6, 7,
                                        8, 9, 10, 11, 12, 13, 14, 15);
      acc1 = __builtin_amdgcn_wmma_f32_16x16x32_bf16(
          false, a, false, b, (short)0, acc1, false, false);
    }
    __syncthreads();
  }

  // epilogue: fused BN (y = g*(x+b)+bt), optional erf-GELU, optional residual
#pragma unroll
  for (int sub = 0; sub < 2; sub++) {
    const int gn = n0 + wc * 32 + sub * 16 + l16;
    if (gn >= N) continue;
    float sc = 1.0f, sh = 0.0f;
    if (gamma) { sc = gamma[gn]; sh = sc * bias[gn] + beta[gn]; }
    const v8f acc = sub ? acc1 : acc0;
#pragma unroll
    for (int v = 0; v < 8; v++) {
      const int gm = m0 + wr * 16 + v + 8 * half;
      if (gm < M) {
        float val = acc[v] * sc + sh;
        if (do_gelu) val = gelu_erf(val);
        if (residual) val += residual[(size_t)gm * N + gn];
        C[(size_t)gm * N + gn] = val;
      }
    }
  }
}

// direct 3x3 conv, NHWC, fused BN + optional GELU (stem/downsample: tiny FLOPs)
__global__ void conv3x3_bn_kernel(
    const float* __restrict__ X, const float* __restrict__ Wt,
    const float* __restrict__ bias, const float* __restrict__ gamma,
    const float* __restrict__ beta, float* __restrict__ Y,
    int Bn, int Hin, int Win, int Cin, int Hout, int Wout, int Cout,
    int stride, int do_gelu)
{
  int i = blockIdx.x * blockDim.x + threadIdx.x;
  int total = Bn * Hout * Wout * Cout;
  if (i >= total) return;
  int co = i % Cout; int t = i / Cout;
  int ox = t % Wout; t /= Wout;
  int oy = t % Hout; int b = t / Hout;
  float acc = 0.0f;
  for (int ky = 0; ky < 3; ky++) {
    int iy = oy * stride + ky - 1;
    if (iy < 0 || iy >= Hin) continue;
    for (int kx = 0; kx < 3; kx++) {
      int ix = ox * stride + kx - 1;
      if (ix < 0 || ix >= Win) continue;
      const float* xp = X + ((size_t)(b * Hin + iy) * Win + ix) * Cin;
      const float* wp = Wt + ((size_t)co * Cin * 9) + (ky * 3 + kx);
      for (int ci = 0; ci < Cin; ci++) acc += xp[ci] * wp[(size_t)ci * 9];
    }
  }
  float val = gamma[co] * (acc + bias[co]) + beta[co];
  if (do_gelu) val = gelu_erf(val);
  Y[i] = val;
}

__global__ void nchw2nhwc_kernel(const float* __restrict__ X, float* __restrict__ Y,
                                 int Bn, int C, int H, int W) {
  int i = blockIdx.x * blockDim.x + threadIdx.x;
  int total = Bn * C * H * W;
  if (i >= total) return;
  int w = i % W; int t = i / W;
  int h = t % H; t /= H;
  int c = t % C; int b = t / C;
  Y[((size_t)(b * H + h) * W + w) * C + c] = X[i];
}

__global__ void nhwc2nchw_kernel(const float* __restrict__ X, float* __restrict__ Y,
                                 int Bn, int C, int H, int W) {
  int i = blockIdx.x * blockDim.x + threadIdx.x;
  int total = Bn * C * H * W;
  if (i >= total) return;
  int w = i % W; int t = i / W;
  int h = t % H; t /= H;
  int c = t % C; int b = t / C;
  Y[i] = X[((size_t)(b * H + h) * W + w) * C + c];
}

// DETR-style 2D sine pos-embed, added in place (NHWC)
__global__ void posembed_add_kernel(float* __restrict__ X, int Bn, int H, int W, int C) {
  int i = blockIdx.x * blockDim.x + threadIdx.x;
  int total = Bn * H * W * C;
  if (i >= total) return;
  int c = i % C; int t = i / C;
  int w = t % W; t /= W;
  int h = t % H;
  int npf = C / 2;
  int axy = (c < npf);
  int d = axy ? c : c - npf;
  float pos = (axy ? (float)(h + 1) / (float)H : (float)(w + 1) / (float)W)
              * 6.2831853071795864f;
  float dimt = powf(10000.0f, 2.0f * (float)(d / 2) / (float)npf);
  float arg = pos / dimt;
  X[i] += (d & 1) ? cosf(arg) : sinf(arg);
}

__global__ void l2norm_rows_kernel(const float* __restrict__ X, float* __restrict__ Y,
                                   int Rows, int C) {
  int r = blockIdx.x * blockDim.x + threadIdx.x;
  if (r >= Rows) return;
  const float* xp = X + (size_t)r * C;
  float s = 0.0f;
  for (int c = 0; c < C; c++) s += xp[c] * xp[c];
  float inv = 1.0f / (sqrtf(s) + 1e-12f);
  float* yp = Y + (size_t)r * C;
  for (int c = 0; c < C; c++) yp[c] = xp[c] * inv;
}

// per-row top-(k*dil) of similarity; keep every dil-th index (9 neighbors)
__global__ void topk_kernel(const float* __restrict__ S, int N, int kk, int dil,
                            int* __restrict__ idx_out) {
  int r = blockIdx.x * blockDim.x + threadIdx.x;
  if (r >= N) return;
  float bv[48];
  int   bi[48];
  for (int t = 0; t < kk; t++) { bv[t] = -3.0e38f; bi[t] = 0; }
  const float* row = S + (size_t)r * N;
  for (int j = 0; j < N; j++) {
    float s = row[j];
    if (s > bv[kk - 1]) {
      int p = kk - 1;
      while (p > 0 && bv[p - 1] < s) { bv[p] = bv[p - 1]; bi[p] = bi[p - 1]; p--; }
      bv[p] = s; bi[p] = j;
    }
  }
  for (int t = 0; t < 9; t++) idx_out[(size_t)r * 9 + t] = bi[t * dil];
}

// MRConv: m = max_k(ft[idx[k]] - ft); comb channels interleaved [f0,m0,f1,m1,...]
__global__ void mrconv_kernel(const float* __restrict__ ft, const int* __restrict__ idx,
                              float* __restrict__ comb, int Bn, int N, int C) {
  int i = blockIdx.x * blockDim.x + threadIdx.x;
  int total = Bn * N * C;
  if (i >= total) return;
  int c = i % C; int t = i / C;
  int n = t % N; int b = t / N;
  size_t base = (size_t)(b * N + n);
  float f = ft[base * C + c];
  float m = -3.0e38f;
  const int* ip = idx + base * 9;
  for (int t2 = 0; t2 < 9; t2++) {
    int j = ip[t2];
    float v = ft[((size_t)(b * N + j)) * C + c] - f;
    m = fmaxf(m, v);
  }
  comb[base * (2 * C) + 2 * c]     = f;
  comb[base * (2 * C) + 2 * c + 1] = m;
}

// ---------------------------------------------------------------------------
// Host orchestration
// ---------------------------------------------------------------------------
struct CBN { const float *w, *b, *g, *bt; };

static inline CBN take_cbn(void* const* d_in, int& cur) {
  CBN p;
  p.w  = (const float*)d_in[cur + 0];
  p.b  = (const float*)d_in[cur + 1];
  p.g  = (const float*)d_in[cur + 2];
  p.bt = (const float*)d_in[cur + 3];
  cur += 4;
  return p;
}

extern "C" void kernel_launch(void* const* d_in, const int* in_sizes, int n_in,
                              void* d_out, int out_size, void* d_ws, size_t ws_size,
                              hipStream_t stream) {
  (void)in_sizes; (void)n_in; (void)out_size; (void)ws_size;
  static const int BLK[4] = {2, 2, 6, 2};
  static const int CH[4]  = {48, 96, 240, 384};
  const int B2 = 2;

  // workspace partition (floats)
  const size_t BIG = 1700000;
  float* bufA = (float*)d_ws;          // persistent activation x (NHWC)
  float* bufB = bufA + BIG;
  float* bufC = bufB + BIG;
  float* bufD = bufC + BIG;
  int*   idxb = (int*)(bufD + BIG);    // (B, N, 9)
  float* Sbuf = (float*)(idxb + (size_t)2 * 4096 * 9);   // N x N, reused per batch

  auto ew_grid = [](int total) { return dim3((total + 255) / 256); };

  auto gemm = [&](const float* A, const float* Bm, float* Cm, int M, int N, int K,
                  const CBN* bn, const float* res, int gelu) {
    dim3 grid((M + TBM - 1) / TBM, (N + TBN - 1) / TBN);
    gemm_wmma_bn_kernel<<<grid, 256, 0, stream>>>(
        A, Bm, Cm, M, N, K,
        bn ? bn->b : nullptr, bn ? bn->g : nullptr, bn ? bn->bt : nullptr,
        res, gelu);
  };
  auto conv3 = [&](const float* X, const CBN& p, float* Y, int Hin, int Win,
                   int Cin, int Hout, int Wout, int Cout, int stride, int gelu) {
    int total = B2 * Hout * Wout * Cout;
    conv3x3_bn_kernel<<<ew_grid(total), 256, 0, stream>>>(
        X, p.w, p.b, p.g, p.bt, Y, B2, Hin, Win, Cin, Hout, Wout, Cout, stride, gelu);
  };

  int cur = 0;
  const float* x_in = (const float*)d_in[cur++];     // (2,3,256,256) NCHW

  CBN stem0 = take_cbn(d_in, cur);
  CBN stem1 = take_cbn(d_in, cur);
  CBN stem2 = take_cbn(d_in, cur);

  // ---- stem ----
  nchw2nhwc_kernel<<<ew_grid(B2 * 3 * 256 * 256), 256, 0, stream>>>(x_in, bufB, B2, 3, 256, 256);
  conv3(bufB, stem0, bufC, 256, 256, 3, 128, 128, 24, 2, 1);
  conv3(bufC, stem1, bufB, 128, 128, 24, 64, 64, 48, 2, 1);
  conv3(bufB, stem2, bufA, 64, 64, 48, 64, 64, 48, 1, 0);
  posembed_add_kernel<<<ew_grid(B2 * 64 * 64 * 48), 256, 0, stream>>>(bufA, B2, 64, 64, 48);

  int H = 64, Wd = 64;
  int blkidx = 0;

  for (int s = 0; s < 4; s++) {
    int Cc = CH[s];
    if (s > 0) {
      CBN down = take_cbn(d_in, cur);
      int Hin = H, Win = Wd, Cin = CH[s - 1];
      H /= 2; Wd /= 2;
      conv3(bufA, down, bufB, Hin, Win, Cin, H, Wd, Cc, 2, 0);
      hipMemcpyAsync(bufA, bufB, (size_t)B2 * H * Wd * Cc * sizeof(float),
                     hipMemcpyDeviceToDevice, stream);
    }
    int N = H * Wd;
    for (int blk = 0; blk < BLK[s]; blk++) {
      CBN gfc1 = take_cbn(d_in, cur);
      CBN gmr  = take_cbn(d_in, cur);
      CBN gfc2 = take_cbn(d_in, cur);
      CBN ffc1 = take_cbn(d_in, cur);
      CBN ffc2 = take_cbn(d_in, cur);
      int dil = blkidx / 4 + 1; if (dil > 5) dil = 5;

      // -------- Grapher --------
      // ft = BN(fc1(x))        -> bufB  (B*N, C)
      gemm(bufA, gfc1.w, bufB, B2 * N, Cc, Cc, &gfc1, nullptr, 0);
      // nf = normalize(ft)     -> bufC
      l2norm_rows_kernel<<<ew_grid(B2 * N), 256, 0, stream>>>(bufB, bufC, B2 * N, Cc);
      // per image: S = nf nf^T (WMMA GEMM), then dilated top-k
      for (int b = 0; b < B2; b++) {
        const float* nfb = bufC + (size_t)b * N * Cc;
        gemm(nfb, nfb, Sbuf, N, N, Cc, nullptr, nullptr, 0);
        topk_kernel<<<dim3((N + 63) / 64), 64, 0, stream>>>(
            Sbuf, N, 9 * dil, dil, idxb + (size_t)b * N * 9);
      }
      // comb (interleaved [f, maxrel]) -> bufC  (B*N, 2C)
      mrconv_kernel<<<ew_grid(B2 * N * Cc), 256, 0, stream>>>(bufB, idxb, bufC, B2, N, Cc);
      // y = GELU(BN(mr(comb)))  -> bufD  (B*N, 2C)
      gemm(bufC, gmr.w, bufD, B2 * N, 2 * Cc, 2 * Cc, &gmr, nullptr, 1);
      // x = BN(fc2(y)) + x      -> bufB, then back to bufA
      gemm(bufD, gfc2.w, bufB, B2 * N, Cc, 2 * Cc, &gfc2, bufA, 0);
      hipMemcpyAsync(bufA, bufB, (size_t)B2 * N * Cc * sizeof(float),
                     hipMemcpyDeviceToDevice, stream);

      // -------- FFN --------
      gemm(bufA, ffc1.w, bufB, B2 * N, 4 * Cc, Cc, &ffc1, nullptr, 1);
      gemm(bufB, ffc2.w, bufC, B2 * N, Cc, 4 * Cc, &ffc2, bufA, 0);
      hipMemcpyAsync(bufA, bufC, (size_t)B2 * N * Cc * sizeof(float),
                     hipMemcpyDeviceToDevice, stream);

      blkidx++;
    }
  }

  // final: NHWC -> NCHW into d_out  (2, 384, 8, 8)
  nhwc2nchw_kernel<<<ew_grid(B2 * CH[3] * H * Wd), 256, 0, stream>>>(
      bufA, (float*)d_out, B2, CH[3], H, Wd);
}
